// MCA_ED_51685636440608
// MI455X (gfx1250) — matrix operand order repaired and checked by reading, not compiled, not run on hardware
//
#include <hip/hip_runtime.h>
#include <math.h>

// ---------------------------------------------------------------------------
// MCA_ED forward: 6-layer encoder (SA over y) + 6-layer RELAGA decoder over x.
// All matmuls run through one batched WMMA GEMM kernel:
//   f32 global data, f32->bf16 convert while staging into LDS (double-
//   buffered ping-pong), v_wmma_f32_16x16x32_bf16 accumulation in f32.
// Workspace requirement: ~235 MB (carved from d_ws).
// ---------------------------------------------------------------------------

constexpr int kD  = 768;
constexpr int kH  = 12;
constexpr int kDH = 64;
constexpr int kFF = 3072;
constexpr int kNB = 32;
constexpr int kLX = 256;
constexpr int kLY = 128;
constexpr int kL  = 6;
#define NEGV (-1e9f)

typedef __attribute__((ext_vector_type(16))) __bf16 v16bf;
typedef __attribute__((ext_vector_type(4)))  __bf16 v4bf;
typedef __attribute__((ext_vector_type(8)))  float  v8f;

union FragBf { uint4 q[2]; v16bf v; };

#define LDSS 40  // padded LDS row stride (bf16 elems): 80B rows keep 16B align

// ------------------------------ reductions ---------------------------------
__device__ __forceinline__ float wave_sum(float v) {
#pragma unroll
  for (int o = 16; o > 0; o >>= 1) v += __shfl_down(v, o, 32);
  return v;
}
__device__ __forceinline__ float wave_max(float v) {
#pragma unroll
  for (int o = 16; o > 0; o >>= 1) v = fmaxf(v, __shfl_down(v, o, 32));
  return v;
}
__device__ __forceinline__ float block_sum(float v, float* sb) {
  const int lane = threadIdx.x & 31, w = threadIdx.x >> 5, nw = blockDim.x >> 5;
  v = wave_sum(v);
  if (lane == 0) sb[w] = v;
  __syncthreads();
  if (w == 0) {
    float t = (lane < nw) ? sb[lane] : 0.f;
    t = wave_sum(t);
    if (lane == 0) sb[0] = t;
  }
  __syncthreads();
  float r = sb[0];
  __syncthreads();
  return r;
}
__device__ __forceinline__ float block_max(float v, float* sb) {
  const int lane = threadIdx.x & 31, w = threadIdx.x >> 5, nw = blockDim.x >> 5;
  v = wave_max(v);
  if (lane == 0) sb[w] = v;
  __syncthreads();
  if (w == 0) {
    float t = (lane < nw) ? sb[lane] : -3.4e38f;
    t = wave_max(t);
    if (lane == 0) sb[0] = t;
  }
  __syncthreads();
  float r = sb[0];
  __syncthreads();
  return r;
}

// --------------------------- batched WMMA GEMM -----------------------------
// C[zb,zh] = act( alpha * A[zb,zh] @ B[zb,zh](^T) + bias[zh] )
// Block tile 128(M) x 32*FN(N); 8 waves; wave tile 32 x 16*FN = 2xFN WMMA.
// K stepped by 32 with double-buffered LDS: global loads for tile k+1 are
// issued before the WMMA block of tile k, converted/stored after it.
template <int TRANSB, int FN>
__global__ __launch_bounds__(256) void mca_bgemm_wmma(
    float* __restrict__ C, const float* __restrict__ A,
    const float* __restrict__ Bm, const float* __restrict__ bias,
    int K, int lda, int ldb, int ldc,
    long sAb, long sAh, long sBb, long sBh, long sCb, long sCh, long sBiasH,
    int nh, float alpha, int relu) {
  constexpr int BN = 32 * FN;  // block N tile
  __shared__ __align__(16) __bf16 sA[2][128 * LDSS];
  __shared__ __align__(16) __bf16 sB[2][BN * LDSS];

  const int tid  = threadIdx.x;
  const int lane = tid & 31;
  const int wave = tid >> 5;
  const int half = lane >> 4;   // lane group 0..15 vs 16..31
  const int l15  = lane & 15;

  const int zb = blockIdx.z / nh;
  const int zh = blockIdx.z - zb * nh;
  A  += (size_t)zb * sAb + (size_t)zh * sAh;
  Bm += (size_t)zb * sBb + (size_t)zh * sBh;
  C  += (size_t)zb * sCb + (size_t)zh * sCh;
  if (bias) bias += (size_t)zh * sBiasH;

  const int m0 = blockIdx.y << 7;       // M tile base
  const int n0 = blockIdx.x * BN;       // N tile base
  const int wm = (wave >> 1) << 5;      // wave row: 0/32/64/96
  const int wn = (wave & 1) * (16 * FN);  // wave col

  v8f acc[2][FN] = {};

  auto loadA = [&](int k0, float4* ra) {
#pragma unroll
    for (int i = 0; i < 4; ++i) {
      const int e4 = tid + (i << 8);  // 1024 float4 slots: [128 x 32]
      const int r  = e4 >> 3;
      const int c  = (e4 & 7) << 2;
      ra[i] = *(const float4*)(A + (size_t)(m0 + r) * lda + (k0 + c));
    }
  };
  auto storeA = [&](int buf, const float4* ra) {
#pragma unroll
    for (int i = 0; i < 4; ++i) {
      const int e4 = tid + (i << 8);
      const int r  = e4 >> 3;
      const int c  = (e4 & 7) << 2;
      v4bf h;
      h.x = (__bf16)ra[i].x; h.y = (__bf16)ra[i].y;
      h.z = (__bf16)ra[i].z; h.w = (__bf16)ra[i].w;
      *(v4bf*)(&sA[buf][r * LDSS + c]) = h;
    }
  };
  auto loadB = [&](int k0, float4* rb) {
#pragma unroll
    for (int i = 0; i < FN; ++i) {
      const int e4 = tid + (i << 8);  // 256*FN float4 slots: [32 x BN]
      if (TRANSB) {
        const int n = e4 >> 3;        // 0..BN-1
        const int c = (e4 & 7) << 2;  // k 0..28
        rb[i] = *(const float4*)(Bm + (size_t)(n0 + n) * ldb + (k0 + c));
      } else {
        const int k = e4 / (BN / 4);
        const int n = (e4 % (BN / 4)) << 2;
        rb[i] = *(const float4*)(Bm + (size_t)(k0 + k) * ldb + (n0 + n));
      }
    }
  };
  auto storeB = [&](int buf, const float4* rb) {
#pragma unroll
    for (int i = 0; i < FN; ++i) {
      const int e4 = tid + (i << 8);
      if (TRANSB) {
        const int n = e4 >> 3;
        const int c = (e4 & 7) << 2;
        v4bf h;
        h.x = (__bf16)rb[i].x; h.y = (__bf16)rb[i].y;
        h.z = (__bf16)rb[i].z; h.w = (__bf16)rb[i].w;
        *(v4bf*)(&sB[buf][n * LDSS + c]) = h;
      } else {
        const int k = e4 / (BN / 4);
        const int n = (e4 % (BN / 4)) << 2;
        sB[buf][(n + 0) * LDSS + k] = (__bf16)rb[i].x;
        sB[buf][(n + 1) * LDSS + k] = (__bf16)rb[i].y;
        sB[buf][(n + 2) * LDSS + k] = (__bf16)rb[i].z;
        sB[buf][(n + 3) * LDSS + k] = (__bf16)rb[i].w;
      }
    }
  };

  const int nt = K >> 5;  // number of 32-wide K tiles
  float4 ra[4], rb[FN];
  loadA(0, ra);
  loadB(0, rb);
  storeA(0, ra);
  storeB(0, rb);
  __syncthreads();

  for (int kt = 0; kt < nt; ++kt) {
    const int cur = kt & 1;
    const bool more = (kt + 1) < nt;
    if (more) {  // issue next tile's global loads before the WMMA block
      loadA((kt + 1) << 5, ra);
      loadB((kt + 1) << 5, rb);
    }

    // fragment loads per ISA 16-bit A 16x32 / B 32x16 layouts
    FragBf a[2], b[FN];
#pragma unroll
    for (int fm = 0; fm < 2; ++fm) {
      const __bf16* pa = &sA[cur][(wm + fm * 16 + l15) * LDSS + half * 8];
      a[fm].q[0] = *(const uint4*)(pa);       // K 0..7   (or 8..15)
      a[fm].q[1] = *(const uint4*)(pa + 16);  // K 16..23 (or 24..31)
    }
#pragma unroll
    for (int fn = 0; fn < FN; ++fn) {
      const __bf16* pb = &sB[cur][(wn + fn * 16 + l15) * LDSS + half * 8];
      b[fn].q[0] = *(const uint4*)(pb);
      b[fn].q[1] = *(const uint4*)(pb + 16);
    }
#pragma unroll
    for (int fm = 0; fm < 2; ++fm)
#pragma unroll
      for (int fn = 0; fn < FN; ++fn)
        acc[fm][fn] = __builtin_amdgcn_wmma_f32_16x16x32_bf16(
            false, a[fm].v, false, b[fn].v, (short)0, acc[fm][fn], false, false);

    if (more) {  // convert + store into the other LDS buffer
      storeA(1 - cur, ra);
      storeB(1 - cur, rb);
    }
    __syncthreads();
  }

  // ---- store C (16x16 f32 layout: lane 0-15 -> M=v, lane 16-31 -> M=v+8) ----
#pragma unroll
  for (int fm = 0; fm < 2; ++fm) {
#pragma unroll
    for (int fn = 0; fn < FN; ++fn) {
      const int rbase = m0 + wm + fm * 16 + half * 8;
      const int col   = n0 + wn + fn * 16 + l15;
      const float bv  = bias ? bias[col] : 0.f;
#pragma unroll
      for (int v = 0; v < 8; ++v) {
        float val = acc[fm][fn][v] * alpha + bv;
        if (relu) val = fmaxf(val, 0.f);
        C[(size_t)(rbase + v) * ldc + col] = val;
      }
    }
  }
}

// --------------------------- masked row softmax ----------------------------
__global__ __launch_bounds__(256) void mca_softmax_kernel(
    float* __restrict__ s, const unsigned char* __restrict__ mask,
    int n, int rowsPerBatch) {
  __shared__ float sb[8];
  const int row = blockIdx.x;
  float* p = s + (size_t)row * n;
  const unsigned char* mrow = mask + (size_t)(row / rowsPerBatch) * n;
  const int k = threadIdx.x;
  float x = -3.4e38f;
  if (k < n) x = mrow[k] ? NEGV : p[k];
  const float m = block_max(x, sb);
  const float e = (k < n) ? __expf(x - m) : 0.f;
  const float ssum = block_sum(e, sb);
  if (k < n) p[k] = e / ssum;
}

// ----------------------- fused residual + LayerNorm ------------------------
__global__ __launch_bounds__(256) void mca_ln_kernel(
    float* __restrict__ out, const float* __restrict__ x,
    const float* __restrict__ add, const float* __restrict__ g,
    const float* __restrict__ beta) {
  __shared__ float sb[8];
  const int row = blockIdx.x;
  const int tid = threadIdx.x;
  const float* px = x + (size_t)row * kD;
  const float* pa = add + (size_t)row * kD;
  float v[3];
  float s = 0.f;
#pragma unroll
  for (int i = 0; i < 3; ++i) {
    v[i] = px[tid + i * 256] + pa[tid + i * 256];
    s += v[i];
  }
  const float mean = block_sum(s, sb) * (1.f / kD);
  s = 0.f;
#pragma unroll
  for (int i = 0; i < 3; ++i) { float d = v[i] - mean; s += d * d; }
  const float var = block_sum(s, sb) * (1.f / kD);
  const float inv = rsqrtf(var + 1e-6f);
  float* po = out + (size_t)row * kD;
#pragma unroll
  for (int i = 0; i < 3; ++i) {
    const int c = tid + i * 256;
    po[c] = g[c] * (v[i] - mean) * inv + beta[c];
  }
}

// -------------------- bbox cosine-similarity + softmax ---------------------
__global__ __launch_bounds__(256) void mca_sim_kernel(
    float* __restrict__ sim, const float* __restrict__ bbox) {
  __shared__ float sb[8];
  const int b = blockIdx.x / kLX;
  const int i = blockIdx.x - b * kLX;
  const int j = threadIdx.x;  // blockDim == kLX == 256
  const float4 fi = *(const float4*)(bbox + ((size_t)b * kLX + i) * 4);
  const float4 fj = *(const float4*)(bbox + ((size_t)b * kLX + j) * 4);
  const float prod = fi.x * fj.x + fi.y * fj.y + fi.z * fj.z + fi.w * fj.w;
  const float ni = sqrtf(fi.x * fi.x + fi.y * fi.y + fi.z * fi.z + fi.w * fi.w);
  const float nj = sqrtf(fj.x * fj.x + fj.y * fj.y + fj.z * fj.z + fj.w * fj.w);
  const float x = prod / (ni * nj);
  const float m = block_max(x, sb);
  const float e = __expf(x - m);
  const float ssum = block_sum(e, sb);
  sim[((size_t)b * kLX + i) * kLX + j] = e / ssum;
}

// ------------------------------ host helper --------------------------------
static inline void bgemm(hipStream_t st, float* C, const float* A,
                         const float* B, const float* bias, int M, int N, int K,
                         int lda, int ldb, int ldc, long sAb, long sAh,
                         long sBb, long sBh, long sCb, long sCh, long sBiasH,
                         int nh, int nz, float alpha, int transB, int relu) {
  const bool big = (N % 128) == 0;  // use 128x128 block tile when possible
  dim3 grid(N / (big ? 128 : 64), M / 128, nz);
  if (transB) {
    if (big)
      mca_bgemm_wmma<1, 4><<<grid, 256, 0, st>>>(C, A, B, bias, K, lda, ldb,
          ldc, sAb, sAh, sBb, sBh, sCb, sCh, sBiasH, nh, alpha, relu);
    else
      mca_bgemm_wmma<1, 2><<<grid, 256, 0, st>>>(C, A, B, bias, K, lda, ldb,
          ldc, sAb, sAh, sBb, sBh, sCb, sCh, sBiasH, nh, alpha, relu);
  } else {
    if (big)
      mca_bgemm_wmma<0, 4><<<grid, 256, 0, st>>>(C, A, B, bias, K, lda, ldb,
          ldc, sAb, sAh, sBb, sBh, sCb, sCh, sBiasH, nh, alpha, relu);
    else
      mca_bgemm_wmma<0, 2><<<grid, 256, 0, st>>>(C, A, B, bias, K, lda, ldb,
          ldc, sAb, sAh, sBb, sBh, sCb, sCh, sBiasH, nh, alpha, relu);
  }
}

extern "C" void kernel_launch(void* const* d_in, const int* in_sizes, int n_in,
                              void* d_out, int out_size, void* d_ws,
                              size_t ws_size, hipStream_t stream) {
  (void)in_sizes; (void)n_in; (void)out_size; (void)ws_size;
  const float* y_in = (const float*)d_in[0];
  const float* x_in = (const float*)d_in[1];
  const unsigned char* y_mask = (const unsigned char*)d_in[2];
  const unsigned char* x_mask = (const unsigned char*)d_in[3];
  const float* bbox = (const float*)d_in[5];
  const float* enc_attn_w = (const float*)d_in[6];
  const float* enc_attn_b = (const float*)d_in[7];
  const float* enc_ffn_w1 = (const float*)d_in[8];
  const float* enc_ffn_b1 = (const float*)d_in[9];
  const float* enc_ffn_w2 = (const float*)d_in[10];
  const float* enc_ffn_b2 = (const float*)d_in[11];
  const float* enc_ln_g = (const float*)d_in[12];
  const float* enc_ln_b = (const float*)d_in[13];
  const float* dec_attn_w = (const float*)d_in[14];
  const float* dec_attn_b = (const float*)d_in[15];
  const float* dec_ffn_w1 = (const float*)d_in[16];
  const float* dec_ffn_b1 = (const float*)d_in[17];
  const float* dec_ffn_w2 = (const float*)d_in[18];
  const float* dec_ffn_b2 = (const float*)d_in[19];
  const float* dec_ln_g = (const float*)d_in[20];
  const float* dec_ln_b = (const float*)d_in[21];

  float* ycur = (float*)d_out;                  // [NB, LY, D] -> output y
  float* xcur = ycur + (size_t)kNB * kLY * kD;  // [NB, LX, D] -> output x

  float* ws = (float*)d_ws;
  const size_t PS = (size_t)kNB * kLX * kD;  // projection slab stride
  float* P   = ws;                           // q,k,v projections (3*PS)
  float* S0  = P + 3 * PS;                   // scores / ffn hidden
  float* S4  = S0 + (size_t)kNB * kH * kLX * kLX;
  float* S5  = S4 + PS;
  float* SIM = S5 + PS;                      // [NB, LX, LX]

  const long DD = (long)kD * kD;

  hipMemcpyAsync(ycur, y_in, (size_t)kNB * kLY * kD * sizeof(float),
                 hipMemcpyDeviceToDevice, stream);
  hipMemcpyAsync(xcur, x_in, (size_t)kNB * kLX * kD * sizeof(float),
                 hipMemcpyDeviceToDevice, stream);

  mca_sim_kernel<<<kNB * kLX, kLX, 0, stream>>>(SIM, bbox);

  // -------------------------------- encoder --------------------------------
  for (int i = 0; i < kL; ++i) {
    const float* W  = enc_attn_w + (size_t)i * 4 * DD;
    const float* Wb = enc_attn_b + (size_t)i * 4 * kD;
    // fused q,k,v projections: batch over the weight index
    bgemm(stream, P, ycur, W, Wb, kNB * kLY, kD, kD, kD, kD, kD,
          0, 0, 0, DD, 0, (long)PS, kD, 3, 3, 1.f, 0, 0);
    // scores = q @ k^T / sqrt(dh)   (per b,h via strides)
    bgemm(stream, S0, P, P + PS, nullptr, kLY, kLY, kDH, kD, kD, kLY,
          (long)kLY * kD, kDH, (long)kLY * kD, kDH,
          (long)kH * kLY * kLY, (long)kLY * kLY, 0, kH, kNB * kH, 0.125f, 1, 0);
    mca_softmax_kernel<<<kNB * kH * kLY, 256, 0, stream>>>(S0, y_mask, kLY,
                                                           kH * kLY);
    // att @ v -> [NB, LY, D] head-concat layout via strided C
    bgemm(stream, S4, S0, P + 2 * PS, nullptr, kLY, kDH, kLY, kLY, kD, kD,
          (long)kH * kLY * kLY, (long)kLY * kLY, (long)kLY * kD, kDH,
          (long)kLY * kD, kDH, 0, kH, kNB * kH, 1.f, 0, 0);
    // merge projection
    bgemm(stream, S5, S4, W + 3 * DD, Wb + 3 * kD, kNB * kLY, kD, kD, kD, kD,
          kD, 0, 0, 0, 0, 0, 0, 0, 1, 1, 1.f, 0, 0);
    mca_ln_kernel<<<kNB * kLY, 256, 0, stream>>>(
        ycur, ycur, S5, enc_ln_g + (size_t)(i * 2 + 0) * kD,
        enc_ln_b + (size_t)(i * 2 + 0) * kD);
    // ffn
    bgemm(stream, S0, ycur, enc_ffn_w1 + (size_t)i * kD * kFF,
          enc_ffn_b1 + (size_t)i * kFF, kNB * kLY, kFF, kD, kD, kFF, kFF,
          0, 0, 0, 0, 0, 0, 0, 1, 1, 1.f, 0, 1);
    bgemm(stream, S5, S0, enc_ffn_w2 + (size_t)i * kFF * kD,
          enc_ffn_b2 + (size_t)i * kD, kNB * kLY, kD, kFF, kFF, kD, kD,
          0, 0, 0, 0, 0, 0, 0, 1, 1, 1.f, 0, 0);
    mca_ln_kernel<<<kNB * kLY, 256, 0, stream>>>(
        ycur, ycur, S5, enc_ln_g + (size_t)(i * 2 + 1) * kD,
        enc_ln_b + (size_t)(i * 2 + 1) * kD);
  }

  // -------------------------------- decoder --------------------------------
  for (int i = 0; i < kL; ++i) {
    {  // self attention + relational matmul
      const float* W  = dec_attn_w + (size_t)(i * 2 + 0) * 4 * DD;
      const float* Wb = dec_attn_b + (size_t)(i * 2 + 0) * 4 * kD;
      bgemm(stream, P, xcur, W, Wb, kNB * kLX, kD, kD, kD, kD, kD,
            0, 0, 0, DD, 0, (long)PS, kD, 3, 3, 1.f, 0, 0);
      bgemm(stream, S0, P, P + PS, nullptr, kLX, kLX, kDH, kD, kD, kLX,
            (long)kLX * kD, kDH, (long)kLX * kD, kDH,
            (long)kH * kLX * kLX, (long)kLX * kLX, 0, kH, kNB * kH, 0.125f, 1, 0);
      mca_softmax_kernel<<<kNB * kH * kLX, 256, 0, stream>>>(S0, x_mask, kLX,
                                                             kH * kLX);
      bgemm(stream, S4, S0, P + 2 * PS, nullptr, kLX, kDH, kLX, kLX, kD, kD,
            (long)kH * kLX * kLX, (long)kLX * kLX, (long)kLX * kD, kDH,
            (long)kLX * kD, kDH, 0, kH, kNB * kH, 1.f, 0, 0);
      // rel: out = sim @ out  (sim broadcast across heads: sAh = 0)
      bgemm(stream, S5, SIM, S4, nullptr, kLX, kDH, kLX, kLX, kD, kD,
            (long)kLX * kLX, 0, (long)kLX * kD, kDH, (long)kLX * kD, kDH, 0,
            kH, kNB * kH, 1.f, 0, 0);
      bgemm(stream, S4, S5, W + 3 * DD, Wb + 3 * kD, kNB * kLX, kD, kD, kD, kD,
            kD, 0, 0, 0, 0, 0, 0, 0, 1, 1, 1.f, 0, 0);
      mca_ln_kernel<<<kNB * kLX, 256, 0, stream>>>(
          xcur, xcur, S4, dec_ln_g + (size_t)(i * 3 + 0) * kD,
          dec_ln_b + (size_t)(i * 3 + 0) * kD);
    }
    {  // cross attention + relational matmul
      const float* W  = dec_attn_w + (size_t)(i * 2 + 1) * 4 * DD;
      const float* Wb = dec_attn_b + (size_t)(i * 2 + 1) * 4 * kD;
      bgemm(stream, P, xcur, W, Wb, kNB * kLX, kD, kD, kD, kD, kD,
            0, 0, 0, 0, 0, 0, 0, 1, 1, 1.f, 0, 0);  // q from x
      bgemm(stream, P + PS, ycur, W + DD, Wb + kD, kNB * kLY, kD, kD, kD, kD,
            kD, 0, 0, 0, DD, 0, (long)PS, kD, 2, 2, 1.f, 0, 0);  // k,v from y
      bgemm(stream, S0, P, P + PS, nullptr, kLX, kLY, kDH, kD, kD, kLY,
            (long)kLX * kD, kDH, (long)kLY * kD, kDH,
            (long)kH * kLX * kLY, (long)kLX * kLY, 0, kH, kNB * kH, 0.125f, 1, 0);
      mca_softmax_kernel<<<kNB * kH * kLX, 256, 0, stream>>>(S0, y_mask, kLY,
                                                             kH * kLX);
      bgemm(stream, S4, S0, P + 2 * PS, nullptr, kLX, kDH, kLY, kLY, kD, kD,
            (long)kH * kLX * kLY, (long)kLX * kLY, (long)kLY * kD, kDH,
            (long)kLX * kD, kDH, 0, kH, kNB * kH, 1.f, 0, 0);
      bgemm(stream, S5, SIM, S4, nullptr, kLX, kDH, kLX, kLX, kD, kD,
            (long)kLX * kLX, 0, (long)kLX * kD, kDH, (long)kLX * kD, kDH, 0,
            kH, kNB * kH, 1.f, 0, 0);
      bgemm(stream, S4, S5, W + 3 * DD, Wb + 3 * kD, kNB * kLX, kD, kD, kD, kD,
            kD, 0, 0, 0, 0, 0, 0, 0, 1, 1, 1.f, 0, 0);
      mca_ln_kernel<<<kNB * kLX, 256, 0, stream>>>(
          xcur, xcur, S4, dec_ln_g + (size_t)(i * 3 + 1) * kD,
          dec_ln_b + (size_t)(i * 3 + 1) * kD);
    }
    // ffn
    bgemm(stream, S0, xcur, dec_ffn_w1 + (size_t)i * kD * kFF,
          dec_ffn_b1 + (size_t)i * kFF, kNB * kLX, kFF, kD, kD, kFF, kFF,
          0, 0, 0, 0, 0, 0, 0, 1, 1, 1.f, 0, 1);
    bgemm(stream, S5, S0, dec_ffn_w2 + (size_t)i * kFF * kD,
          dec_ffn_b2 + (size_t)i * kD, kNB * kLX, kD, kFF, kFF, kD, kD,
          0, 0, 0, 0, 0, 0, 0, 1, 1, 1.f, 0, 0);
    mca_ln_kernel<<<kNB * kLX, 256, 0, stream>>>(
        xcur, xcur, S5, dec_ln_g + (size_t)(i * 3 + 2) * kD,
        dec_ln_b + (size_t)(i * 3 + 2) * kD);
  }
}
